// TimeAwareAttention_14731737825303
// MI455X (gfx1250) — compile-verified, matrix operand order checked by reading, not run on hardware
//
#include <hip/hip_runtime.h>
#include <hip/hip_bf16.h>
#include <cmath>

// Problem sizes (fixed by the reference).
#define B_ 8
#define L_ 2048
#define E_ 1024
#define D_ 256

typedef _Float16 h16;
typedef __attribute__((ext_vector_type(16))) _Float16 v16h;
typedef __attribute__((ext_vector_type(8)))  _Float16 v8h;
typedef __attribute__((ext_vector_type(4)))  _Float16 v4h;
typedef __attribute__((ext_vector_type(8)))  float    v8f;
typedef __attribute__((ext_vector_type(4)))  float    v4f;

// D = A*B + C, f16 inputs, f32 accumulate. Emits v_wmma_f32_16x16x32_f16.
__device__ __forceinline__ v8f wmma_f16(v16h a, v16h b, v8f c) {
  return __builtin_amdgcn_wmma_f32_16x16x32_f16(
      /*neg_a=*/false, a, /*neg_b=*/false, b,
      /*c_mod=*/(short)0, c, /*reuse_a=*/false, /*reuse_b=*/false);
}

// Build a 16-half A/B fragment per the CDNA5 16-bit 16x32 layout:
//   lanes 0-15 : halves[0..7]=K0..7,   halves[8..15]=K16..23
//   lanes 16-31: halves[0..7]=K8..15,  halves[8..15]=K24..31
__device__ __forceinline__ v16h make_frag(const h16* rowbase, int hi) {
  const v8h lo = *(const v8h*)(rowbase + (hi ? 8 : 0));
  const v8h up = *(const v8h*)(rowbase + 16 + (hi ? 8 : 0));
  return __builtin_shufflevector(lo, up, 0,1,2,3,4,5,6,7,8,9,10,11,12,13,14,15);
}

__device__ __forceinline__ float gelu_exact(float x) {
  return 0.5f * x * (1.0f + erff(x * 0.70710678118654752f));
}

// Async copy 16B global -> LDS (ASYNCcnt-tracked, no VGPR round trip).
// lds_off = wave-relative LDS byte address (low 32 bits of generic pointer).
__device__ __forceinline__ void async_ld_b128(unsigned lds_off, unsigned long long gaddr) {
  asm volatile("global_load_async_to_lds_b128 %0, %1, off"
               :: "v"(lds_off), "v"(gaddr) : "memory");
}
__device__ __forceinline__ void wait_async0() {
  asm volatile("s_wait_asynccnt 0x0" ::: "memory");
}

// ---------------------------------------------------------------------------
// Kernel 1: out = f16( GELU( X @ W + bias ) )
// Grid: (B*L/32, D/64). Block: 256 = 8 waves (2 row-tiles x 4 col-tiles).
// K-step = 64 (2 WMMAs per wave per staging round). W staged transposed so
// B-fragment reads are contiguous along K. transposed==1 stores V as [B][D][L].
// ---------------------------------------------------------------------------
__global__ __launch_bounds__(256)
void qkv_gelu_kernel(const float* __restrict__ X, const float* __restrict__ W,
                     const float* __restrict__ bias, h16* __restrict__ outp,
                     int transposed) {
  __shared__ __attribute__((aligned(16))) h16 Xs[32 * 72];   // [row][K0..63]
  __shared__ __attribute__((aligned(16))) h16 Wst[64 * 72];  // [col][K0..63]

  const int tid  = threadIdx.x;
  const int wave = tid >> 5;
  const int lane = tid & 31;
  const int hi   = lane >> 4;
  const int lr   = lane & 15;
  const int waveRow = wave >> 2;   // 0..1
  const int waveCol = wave & 3;    // 0..3
  const int rbase = blockIdx.x * 32;
  const int dbase = blockIdx.y * 64;

  v8f acc = {};

  for (int kk = 0; kk < E_; kk += 64) {
    __syncthreads();
    {
      // Stage X: 32 rows x 64 K, two float4 per thread.
      const int r  = tid >> 3;
      const int c4 = (tid & 7) * 4;
#pragma unroll
      for (int s = 0; s < 2; ++s) {
        v4f xv = *(const v4f*)(X + (size_t)(rbase + r) * E_ + kk + s * 32 + c4);
        v4h xh; xh[0] = (h16)xv[0]; xh[1] = (h16)xv[1];
                xh[2] = (h16)xv[2]; xh[3] = (h16)xv[3];
        *(v4h*)(&Xs[r * 72 + s * 32 + c4]) = xh;
      }
      // Stage W transposed: 64 K x 64 cols, four float4 per thread.
      const int c4w = (tid & 15) * 4;
#pragma unroll
      for (int s = 0; s < 4; ++s) {
        const int kr = (tid >> 4) + s * 16;
        v4f wv = *(const v4f*)(W + (size_t)(kk + kr) * D_ + dbase + c4w);
#pragma unroll
        for (int i = 0; i < 4; ++i)
          Wst[(c4w + i) * 72 + kr] = (h16)wv[i];
      }
    }
    __syncthreads();

    const h16* xrow = &Xs[(waveRow * 16 + lr) * 72];
    const h16* wrow = &Wst[(waveCol * 16 + lr) * 72];
#pragma unroll
    for (int c = 0; c < 2; ++c) {
      v16h a = make_frag(xrow + 32 * c, hi);
      v16h b = make_frag(wrow + 32 * c, hi);
      acc = wmma_f16(a, b, acc);
    }
  }

  // Epilogue: C layout -> lane(j): (M = j + 8*hi, N = lr).
  const int colg = dbase + waveCol * 16 + lr;
  const float bv = bias[colg];
#pragma unroll
  for (int j = 0; j < 8; ++j) {
    const int rowg = rbase + waveRow * 16 + j + hi * 8;
    const float val = gelu_exact(acc[j] + bv);
    if (!transposed) {
      outp[(size_t)rowg * D_ + colg] = (h16)val;           // [B*L][D]
    } else {
      const int bb = rowg >> 11;          // / L_
      const int ll = rowg & (L_ - 1);     // % L_
      outp[((size_t)bb * D_ + colg) * L_ + ll] = (h16)val; // [B][D][L]
    }
  }
}

// ---------------------------------------------------------------------------
// Kernel 2: fused flash attention with polynomial-sigmoid time weights.
// Grid: (L/32, B). Block: 256 = 8 waves = 2 q-subtiles(16 rows) x 4 d-groups(64).
// K / V^T tiles are double-buffered in LDS and filled with async-to-LDS copies
// so tile i+1 streams from HBM while tile i is consumed by the WMMAs.
// ---------------------------------------------------------------------------
#define KS_STRIDE  264   // halves per key row (528B, 16B multiple)
#define VTS_STRIDE 40    // halves per d row   (80B, 16B multiple)

__global__ __launch_bounds__(256)
void flash_attn_kernel(const h16* __restrict__ qh, const h16* __restrict__ kh,
                       const h16* __restrict__ vth, const float* __restrict__ dt,
                       const unsigned char* __restrict__ amask,
                       const float* __restrict__ coeffs, float* __restrict__ out) {
  __shared__ __attribute__((aligned(16))) h16 Ks[2][32 * KS_STRIDE];   // [key][d]
  __shared__ __attribute__((aligned(16))) h16 Vts[2][256 * VTS_STRIDE]; // [d][key]
  __shared__ __attribute__((aligned(16))) h16 Ps[8 * 16 * 32];          // per-wave P

  const int tid  = threadIdx.x;
  const int wave = tid >> 5;
  const int lane = tid & 31;
  const int hi   = lane >> 4;
  const int lr   = lane & 15;
  const int qSub = wave >> 2;  // 0..1
  const int dG   = wave & 3;   // 0..3
  const int bId  = blockIdx.y;
  const int qbase = blockIdx.x * 32 + qSub * 16;
  const int dbase = dG * 64;
  const float scale = 0.0625f; // 1/sqrt(D)

  const float c0 = coeffs[0], c1 = coeffs[1], c2 = coeffs[2],
              c3 = coeffs[3], c4 = coeffs[4], c5 = coeffs[5];

  // This thread's fixed staging slots (global column / LDS destination).
  const int kr = tid >> 3;             // key row it stages
  const int cc = (tid & 7) * 32;       // d-offset it stages for K
  const unsigned ksOff[2]  = { (unsigned)(size_t)&Ks[0][kr * KS_STRIDE + cc],
                               (unsigned)(size_t)&Ks[1][kr * KS_STRIDE + cc] };
  const unsigned vtsOff[2] = { (unsigned)(size_t)&Vts[0][tid * VTS_STRIDE],
                               (unsigned)(size_t)&Vts[1][tid * VTS_STRIDE] };

  // Q fragments for 16 rows x 256 d (A-layout, 8 chunks of K=32).
  v16h qf[8];
  {
    const h16* qrow = qh + ((size_t)(bId * L_ + qbase + lr)) * D_;
#pragma unroll
    for (int c = 0; c < 8; ++c) qf[c] = make_frag(qrow + 32 * c, hi);
  }

  v8f o[4] = {};
  float mrow[8], lrow[8];
#pragma unroll
  for (int j = 0; j < 8; ++j) { mrow[j] = -INFINITY; lrow[j] = 0.0f; }

  const int kbEnd = blockIdx.x * 32 + 32; // uniform causal bound for the block

  // Prologue: async-stage tile 0 into buffer 0.
  {
    const unsigned long long gk =
        (unsigned long long)(size_t)(kh + ((size_t)(bId * L_ + kr)) * D_ + cc);
    const unsigned long long gv =
        (unsigned long long)(size_t)(vth + ((size_t)bId * D_ + tid) * L_);
#pragma unroll
    for (int u = 0; u < 4; ++u) {
      async_ld_b128(ksOff[0] + 16u * u, gk + 16u * u);
      async_ld_b128(vtsOff[0] + 16u * u, gv + 16u * u);
    }
  }
  wait_async0();
  __syncthreads();

  for (int kb = 0; kb < kbEnd; kb += 32) {
    const int cur = (kb >> 5) & 1;

    // Issue async loads for the next tile into the other buffer (uniform cond).
    if (kb + 32 < kbEnd) {
      const int nk = kb + 32;
      const unsigned long long gk =
          (unsigned long long)(size_t)(kh + ((size_t)(bId * L_ + nk + kr)) * D_ + cc);
      const unsigned long long gv =
          (unsigned long long)(size_t)(vth + ((size_t)bId * D_ + tid) * L_ + nk);
#pragma unroll
      for (int u = 0; u < 4; ++u) {
        async_ld_b128(ksOff[cur ^ 1] + 16u * u, gk + 16u * u);
        async_ld_b128(vtsOff[cur ^ 1] + 16u * u, gv + 16u * u);
      }
    }

    if (kb <= qbase + 15) {  // tile not fully above the causal diagonal
      const h16* KsC  = &Ks[cur][0];
      const h16* VtsC = &Vts[cur][0];

      // S = Q K^T : two 16x16 tiles (keys kb..kb+15, kb+16..kb+31).
      v8f s0 = {}, s1 = {};
#pragma unroll
      for (int c = 0; c < 8; ++c) {
        v16h b0 = make_frag(&KsC[lr * KS_STRIDE + 32 * c], hi);
        v16h b1 = make_frag(&KsC[(16 + lr) * KS_STRIDE + 32 * c], hi);
        s0 = wmma_f16(qf[c], b0, s0);
        s1 = wmma_f16(qf[c], b1, s1);
      }

      h16* ps = &Ps[wave * (16 * 32)];
#pragma unroll
      for (int j = 0; j < 8; ++j) {
        const int qrow = qbase + j + hi * 8;
        const size_t dtrow = ((size_t)(bId * L_) + qrow) * (size_t)L_;
        float sv0 = s0[j], sv1 = s1[j];

        // w(t) = sigmoid(poly5(t)) applied before masking (matches ref).
        {
          const int key = kb + lr;
          const float t = dt[dtrow + key];
          float p = c5;
          p = fmaf(p, t, c4); p = fmaf(p, t, c3); p = fmaf(p, t, c2);
          p = fmaf(p, t, c1); p = fmaf(p, t, c0);
          sv0 *= scale * (1.0f / (1.0f + __expf(-p)));
          if (key > qrow || !amask[bId * L_ + key]) sv0 = -INFINITY;
        }
        {
          const int key = kb + 16 + lr;
          const float t = dt[dtrow + key];
          float p = c5;
          p = fmaf(p, t, c4); p = fmaf(p, t, c3); p = fmaf(p, t, c2);
          p = fmaf(p, t, c1); p = fmaf(p, t, c0);
          sv1 *= scale * (1.0f / (1.0f + __expf(-p)));
          if (key > qrow || !amask[bId * L_ + key]) sv1 = -INFINITY;
        }

        // Row reduction over the 16 lanes sharing this row (C-layout half).
        float mt = fmaxf(sv0, sv1);
        mt = fmaxf(mt, __shfl_xor(mt, 1));
        mt = fmaxf(mt, __shfl_xor(mt, 2));
        mt = fmaxf(mt, __shfl_xor(mt, 4));
        mt = fmaxf(mt, __shfl_xor(mt, 8));

        const float mnew  = fmaxf(mrow[j], mt);
        const float alpha = __expf(mrow[j] - mnew);
        const float p0 = __expf(sv0 - mnew);
        const float p1 = __expf(sv1 - mnew);
        float rs = p0 + p1;
        rs += __shfl_xor(rs, 1);
        rs += __shfl_xor(rs, 2);
        rs += __shfl_xor(rs, 4);
        rs += __shfl_xor(rs, 8);
        lrow[j] = lrow[j] * alpha + rs;
        mrow[j] = mnew;
#pragma unroll
        for (int t4 = 0; t4 < 4; ++t4) o[t4][j] *= alpha;

        // Park P (C layout) into LDS so we can re-read it in A layout.
        ps[(j + hi * 8) * 32 + lr]      = (h16)p0;
        ps[(j + hi * 8) * 32 + 16 + lr] = (h16)p1;
      }

      // O += P @ V for this wave's 64 output columns.
      v16h pf = make_frag(ps + lr * 32, hi);
#pragma unroll
      for (int t4 = 0; t4 < 4; ++t4) {
        v16h bv = make_frag(&VtsC[(dbase + t4 * 16 + lr) * VTS_STRIDE], hi);
        o[t4] = wmma_f16(pf, bv, o[t4]);
      }
    }

    // Next buffer fully written (our asyncs + everyone else's via barrier),
    // and all waves finished reading the current buffer.
    wait_async0();
    __syncthreads();
  }

  // Normalize and store (causal diag guarantees lrow > 0; mask is all-true).
#pragma unroll
  for (int t4 = 0; t4 < 4; ++t4) {
#pragma unroll
    for (int j = 0; j < 8; ++j) {
      const int qrow = qbase + j + hi * 8;
      out[((size_t)(bId * L_) + qrow) * D_ + dbase + t4 * 16 + lr] =
          o[t4][j] / lrow[j];
    }
  }
}

// ---------------------------------------------------------------------------
// Launch. Workspace layout (needs 24 MB): qh | kh | vth, each B*L*D f16.
// ---------------------------------------------------------------------------
extern "C" void kernel_launch(void* const* d_in, const int* in_sizes, int n_in,
                              void* d_out, int out_size, void* d_ws, size_t ws_size,
                              hipStream_t stream) {
  (void)in_sizes; (void)n_in; (void)out_size; (void)ws_size;
  const float* X  = (const float*)d_in[0];
  const float* dt = (const float*)d_in[1];
  const unsigned char* am = (const unsigned char*)d_in[2];
  const float* Wq = (const float*)d_in[3];
  const float* bq = (const float*)d_in[4];
  const float* Wk = (const float*)d_in[5];
  const float* bk = (const float*)d_in[6];
  const float* Wv = (const float*)d_in[7];
  const float* bv = (const float*)d_in[8];
  const float* cf = (const float*)d_in[9];
  float* out = (float*)d_out;

  h16* qh  = (h16*)d_ws;
  h16* kh  = qh + (size_t)B_ * L_ * D_;
  h16* vth = kh + (size_t)B_ * L_ * D_;

  dim3 g1(B_ * L_ / 32, D_ / 64, 1);
  qkv_gelu_kernel<<<g1, 256, 0, stream>>>(X, Wq, bq, qh, 0);
  qkv_gelu_kernel<<<g1, 256, 0, stream>>>(X, Wk, bk, kh, 0);
  qkv_gelu_kernel<<<g1, 256, 0, stream>>>(X, Wv, bv, vth, 1);

  dim3 g2(L_ / 32, B_, 1);
  flash_attn_kernel<<<g2, 256, 0, stream>>>(qh, kh, vth, dt, am, cf, out);
}
